// VectorQuantizer_22703197126927
// MI455X (gfx1250) — matrix operand order repaired and checked by reading, not compiled, not run on hardware
//
#include <hip/hip_runtime.h>

#define K_EMB 4096
#define D_EMB 64
#define K_PAD (K_EMB + 16)   // one extra chunk so the pipelined loads never branch

typedef __attribute__((ext_vector_type(16))) _Float16 v16h;
typedef __attribute__((ext_vector_type(8)))  float    v8f;

// ---------------------------------------------------------------------------
// Kernel 1: convert embeddings fp32 -> f16 staging, compute h[c] = 0.5*||e_c||^2
// grid: K_EMB blocks x 32 threads (one wave per embedding)
// ---------------------------------------------------------------------------
__global__ void vq_prep(const float* __restrict__ emb,
                        _Float16* __restrict__ E16,
                        float* __restrict__ h) {
  const int c = blockIdx.x;
  const int l = threadIdx.x;            // 0..31
  const float* er = emb + (size_t)c * D_EMB;
  float x0 = er[l];
  float x1 = er[l + 32];
  E16[(size_t)c * D_EMB + l]      = (_Float16)x0;
  E16[(size_t)c * D_EMB + l + 32] = (_Float16)x1;
  float s = x0 * x0 + x1 * x1;
  #pragma unroll
  for (int m = 16; m >= 1; m >>= 1) s += __shfl_xor(s, m, 32);
  if (l == 0) h[c] = 0.5f * s;
}

// ---------------------------------------------------------------------------
// Kernel 2: WMMA scoring + argmax.  One wave (32 lanes) owns 16 input rows and
// sweeps all kEmb embeddings in 16-column chunks.  score = x.e - 0.5*||e||^2
// (C accumulator initialized to -0.5*||e||^2, so WMMA output == score).
// Two-deep software pipeline:
//   - B operands / h for chunk c0+16 are loaded while chunk c0 computes
//   - argmax selection runs one chunk behind the WMMAs, hiding the
//     WMMA(f16)->VALU accumulator hazard with real work
// kEmb is a RUNTIME argument so the trip count is unknown -> the unroller
// must honor '#pragma unroll 2' (a constant bound triggered a full 255-way
// unroll: 11k instructions, register spills, rematerialized cvts).
// ---------------------------------------------------------------------------
__global__ void __launch_bounds__(256) vq_argmin(
    const float* __restrict__ X,
    const _Float16* __restrict__ E16,
    const float* __restrict__ h,
    float* __restrict__ out_idx,   // N floats (indices stored as float)
    int N, int kEmb) {
  const int lane    = threadIdx.x & 31;
  const int wave    = blockIdx.x * (blockDim.x >> 5) + (threadIdx.x >> 5);
  const int rowBase = wave * 16;
  if (rowBase >= N) return;                 // wave-uniform; EXEC stays full

  const int halfSel = lane >> 4;            // 0 for lanes 0-15, 1 for 16-31
  const int hi      = halfSel * 8;          // A-operand K sub-offset
  const int colLane = lane & 15;
  const int row     = rowBase + colLane;    // A: lane holds row l%16

  // --- A operands (convert fp32 -> f16 on the fly; reused for all K) ------
  // A k order used here (same as B): halves 0..15 = K (2*hi..2*hi+15), contiguous per lane half
  const float* xr = X + (size_t)row * D_EMB;
  v16h a_lo, a_hi;
  #pragma unroll
  for (int i = 0; i < 8; ++i) {
    a_lo[i]     = (_Float16)xr[2 * hi + i];
    a_lo[i + 8] = (_Float16)xr[8 + 2 * hi + i];
    a_hi[i]     = (_Float16)xr[32 + 2 * hi + i];
    a_hi[i + 8] = (_Float16)xr[40 + 2 * hi + i];
  }

  float best[8];
  int   bidx[8];
  #pragma unroll
  for (int r = 0; r < 8; ++r) { best[r] = -3.0e38f; bidx[r] = 0; }

  // --- pipelined K loop ----------------------------------------------------
  const int CH = 16 * D_EMB;                        // halves per 16-col chunk
  const _Float16* bp = E16 + (size_t)colLane * D_EMB + halfSel * 16;

  // B k order used here (same as A): halves 0..15 = K (0..15 | 16..31) per lane half
  v16h bl = *(const v16h*)bp;                       // dims  0..31, chunk 0
  v16h bh = *(const v16h*)(bp + 32);                // dims 32..63, chunk 0
  float nh = -h[colLane];
  bp += CH;

  // prologue: score chunk 0, start loads for chunk 1
  v8f accp;
  {
    const v16h cbl = bl;
    const v16h cbh = bh;
    const float cnh = nh;
    bl = *(const v16h*)bp;
    bh = *(const v16h*)(bp + 32);
    nh = -h[16 + colLane];
    bp += CH;
    accp = (v8f){ cnh, cnh, cnh, cnh, cnh, cnh, cnh, cnh };
    accp = __builtin_amdgcn_wmma_f32_16x16x32_f16(false, a_lo, false, cbl,
                                                  (short)0, accp, false, false);
    accp = __builtin_amdgcn_wmma_f32_16x16x32_f16(false, a_hi, false, cbh,
                                                  (short)0, accp, false, false);
  }

  #pragma unroll 2
  for (int c0 = 16; c0 < kEmb; c0 += 16) {
    const v16h cbl = bl;
    const v16h cbh = bh;
    const float cnh = nh;

    // issue next chunk's loads (padded staging: always in-bounds)
    bl = *(const v16h*)bp;
    bh = *(const v16h*)(bp + 32);
    nh = -h[c0 + 16 + colLane];
    __builtin_prefetch((const void*)(bp + CH), 0, 3);
    bp += CH;

    // WMMAs for the current chunk
    v8f acc = { cnh, cnh, cnh, cnh, cnh, cnh, cnh, cnh };
    acc = __builtin_amdgcn_wmma_f32_16x16x32_f16(false, a_lo, false, cbl,
                                                 (short)0, acc, false, false);
    acc = __builtin_amdgcn_wmma_f32_16x16x32_f16(false, a_hi, false, cbh,
                                                 (short)0, acc, false, false);

    // selection for the PREVIOUS chunk (hides the WMMA->VALU hazard);
    // strict '>' keeps lowest column index on ties
    const int pcol = c0 - 16 + colLane;
    #pragma unroll
    for (int r = 0; r < 8; ++r) {
      const float s = accp[r];
      const int t = s > best[r];
      best[r] = t ? s    : best[r];
      bidx[r] = t ? pcol : bidx[r];
    }
    accp = acc;
  }

  // epilogue: select the final chunk
  {
    const int pcol = kEmb - 16 + colLane;
    #pragma unroll
    for (int r = 0; r < 8; ++r) {
      const float s = accp[r];
      const int t = s > best[r];
      best[r] = t ? s    : best[r];
      bidx[r] = t ? pcol : bidx[r];
    }
  }

  // Reduce the 16 column slots (lanes 0-15 and 16-31 independently; butterfly
  // masks < 16 never cross the half boundary).  Branch-free tie-break: prefer
  // lower index on equal scores (matches argmin's first-hit semantics).
  #pragma unroll
  for (int r = 0; r < 8; ++r) {
    float v  = best[r];
    int   ix = bidx[r];
    #pragma unroll
    for (int m = 1; m < 16; m <<= 1) {
      const float ov = __shfl_xor(v, m, 32);
      const int   oi = __shfl_xor(ix, m, 32);
      const int take = (int)(ov > v) | ((int)(ov == v) & (int)(oi < ix));
      v  = take ? ov : v;
      ix = take ? oi : ix;
    }
    best[r] = v;
    bidx[r] = ix;
  }

  // C layout: VGPR r holds row r (lanes 0-15) / row r+8 (lanes 16-31)
  if (colLane == 0) {
    const int rbase = rowBase + halfSel * 8;
    #pragma unroll
    for (int r = 0; r < 8; ++r) out_idx[rbase + r] = (float)bidx[r];
  }
}

// ---------------------------------------------------------------------------
// Kernel 3: gather fp32 codebook rows into d_out (quantized_st forward value)
// and produce per-block squared-error partial sums (deterministic tree).
// ---------------------------------------------------------------------------
__global__ void __launch_bounds__(256) vq_gather(
    const float* __restrict__ X,
    const float* __restrict__ emb,
    const float* __restrict__ out_idx,
    float* __restrict__ out_q,
    float* __restrict__ partials,
    int total) {
  __shared__ float sm[256];
  const int gid = blockIdx.x * blockDim.x + threadIdx.x;
  float local = 0.0f;
  if (gid < total) {
    const int n = gid >> 6;
    const int d = gid & 63;
    const int idx = (int)out_idx[n];
    const float q = emb[(size_t)idx * D_EMB + d];
    const float x = X[gid];
    out_q[gid] = q;
    const float df = q - x;
    local = df * df;
  }
  sm[threadIdx.x] = local;
  __syncthreads();
  #pragma unroll
  for (int s = 128; s > 0; s >>= 1) {
    if (threadIdx.x < s) sm[threadIdx.x] += sm[threadIdx.x + s];
    __syncthreads();
  }
  if (threadIdx.x == 0) partials[blockIdx.x] = sm[0];
}

// ---------------------------------------------------------------------------
// Kernel 4: final deterministic reduction -> loss = 1.25 * mean(diff^2)
// ---------------------------------------------------------------------------
__global__ void __launch_bounds__(256) vq_loss(
    const float* __restrict__ partials, int nparts,
    float* __restrict__ loss, float scale) {
  __shared__ float sm[256];
  float s = 0.0f;
  for (int i = threadIdx.x; i < nparts; i += 256) s += partials[i];
  sm[threadIdx.x] = s;
  __syncthreads();
  #pragma unroll
  for (int k = 128; k > 0; k >>= 1) {
    if (threadIdx.x < k) sm[threadIdx.x] += sm[threadIdx.x + k];
    __syncthreads();
  }
  if (threadIdx.x == 0) *loss = sm[0] * scale;
}

// ---------------------------------------------------------------------------
extern "C" void kernel_launch(void* const* d_in, const int* in_sizes, int n_in,
                              void* d_out, int out_size, void* d_ws, size_t ws_size,
                              hipStream_t stream) {
  const float* X   = (const float*)d_in[0];   // [32,32,32,64] fp32
  const float* emb = (const float*)d_in[1];   // [4096,64] fp32

  const int total = in_sizes[0];              // 2,097,152 elements
  const int N     = total / D_EMB;            // 32,768 rows

  float* out      = (float*)d_out;
  float* out_q    = out;                      // [total] quantized_st
  float* out_loss = out + total;              // [1]
  float* out_idx  = out + total + 1;          // [N] indices (as float)

  // workspace layout (all 16B aligned):
  //   E16     : (K_EMB+16)*64 f16   = 526,336 B   (padded one chunk)
  //   h       : (K_EMB+16)   f32    =  16,448 B   (padded one chunk)
  //   partials: 8192 f32            =  32,768 B
  char* ws = (char*)d_ws;
  _Float16* E16     = (_Float16*)ws;
  float*    h       = (float*)(ws + (size_t)K_PAD * D_EMB * 2);
  float*    partial = (float*)(ws + (size_t)K_PAD * D_EMB * 2
                                  + (size_t)K_PAD * 4 + 48 /*align to 16*/);

  // 1) stage f16 codebook + half squared norms
  vq_prep<<<K_EMB, 32, 0, stream>>>(emb, E16, h);

  // 2) WMMA scoring + argmin   (one wave per 16 rows, 8 waves/block)
  const int waves  = N / 16;                  // 2048
  const int blocks = (waves + 7) / 8;         // 256
  vq_argmin<<<blocks, 256, 0, stream>>>(X, E16, h, out_idx, N, K_EMB);

  // 3) gather + partial sums
  const int gblocks = (total + 255) / 256;    // 8192
  vq_gather<<<gblocks, 256, 0, stream>>>(X, emb, out_idx, out_q, partial, total);

  // 4) loss
  vq_loss<<<1, 256, 0, stream>>>(partial, gblocks, out_loss,
                                 1.25f / (float)total);
}
